// MambaG2G_6536940224929
// MI455X (gfx1250) — matrix operand order repaired and checked by reading, not compiled, not run on hardware
//
#include <hip/hip_runtime.h>
#include <cstdint>
#include <cstddef>

// ---------------- problem constants ----------------
#define D_MODEL   1000
#define D_MODEL_P 1024   // padded K for GEMM1
#define D_INNER   2000
#define D_INNER_P 2048   // padded K for GEMM4
#define D_STATE   8
#define D_CONV    4
#define DT_RANK   63
#define DBC_W     (DT_RANK + 2*D_STATE)   // 79
#define DIM_OUT   64
#define BATCH     8
#define SEQ       2048
#define MROWS     (BATCH*SEQ)             // 16384
#define N_IN      (2*D_INNER)             // 4000
#define N_IN_P    4032                    // padded N for GEMM1 (63 * 64)
#define N_OUT_P   1024                    // padded N for GEMM4 (16 * 64)

typedef unsigned short u16;
typedef __attribute__((ext_vector_type(16))) __bf16 v16bf;
typedef __attribute__((ext_vector_type(8)))  float  v8f;

// ---------------- scalar helpers ----------------
__device__ __forceinline__ u16 f2bf(float f) {
  union { float f; uint32_t u; } v; v.f = f;
  uint32_t r = v.u + 0x7fffu + ((v.u >> 16) & 1u);   // round-nearest-even
  return (u16)(r >> 16);
}
__device__ __forceinline__ float bf2f(u16 h) {
  union { uint32_t u; float f; } v; v.u = ((uint32_t)h) << 16;
  return v.f;
}
__device__ __forceinline__ float sigmoidf_(float x){ return 1.f/(1.f+__expf(-x)); }
__device__ __forceinline__ float siluf_(float x){ return x*sigmoidf_(x); }
__device__ __forceinline__ float softplusf_(float x){ return x>20.f ? x : log1pf(__expf(x)); }
__device__ __forceinline__ float eluf_(float x){ return x>0.f ? x : (__expf(x)-1.f); }

union FragAB { uint4 q[2]; v16bf v; };

// CDNA5 async global->LDS copy (ASYNCcnt path), 16 bytes per lane.
// lds_off = byte offset within workgroup LDS allocation (low 32 bits of generic ptr).
__device__ __forceinline__ void async_copy_b128(uint32_t lds_off, const void* gptr) {
  asm volatile("global_load_async_to_lds_b128 %0, %1, off"
               :: "v"(lds_off), "v"((uint64_t)(uintptr_t)gptr) : "memory");
}
__device__ __forceinline__ void wait_asynccnt0() {
  asm volatile("s_wait_asynccnt 0x0" ::: "memory");
}

// ---------------- bf16 WMMA GEMM: C[M,Np] = A[M,K]bf16 @ B[K,Np]bf16 ----------------
// block tile 128x64, 8 waves (wave32) in 4x2, each wave 32x32 via 2x2 wmma 16x16x32.
// All dims pre-padded: K,lda mult of 32; ldb (=ldc) mult of 64; M mult of 128. No guards.
template <bool OUT_BF16>
__global__ __launch_bounds__(256)
void k_gemm_bf16(const u16* __restrict__ A, const u16* __restrict__ B,
                 void* __restrict__ Cout, int K, int lda, int ldb) {
  __shared__ __align__(16) u16 As[128*40];   // 128 rows, stride 40 halves (16B aligned rows)
  __shared__ __align__(16) u16 Bs[64*40];    // transposed: Bs[n*40 + k]
  const int tid    = threadIdx.x;
  const int lane   = tid & 31;
  const int wave   = tid >> 5;
  const int wm     = wave >> 1;              // 0..3
  const int wn     = wave & 1;               // 0..1
  const int bm     = blockIdx.y * 128;
  const int bn     = blockIdx.x * 64;
  const int lane16 = lane & 15;
  const int lh     = lane >> 4;

  v8f acc[2][2] = {};

  const int arow  = tid >> 1;                // 0..127
  const int ahalf = tid & 1;                 // which 16-element half of the 32-wide row
  const uint32_t aldsoff = (uint32_t)(uintptr_t)&As[arow*40 + 16*ahalf];

  // B-tile cooperative mapping: each thread owns 8 contiguous bf16 along n of one k-row
  const int bk  = tid >> 3;                  // 0..31
  const int bnn = (tid & 7) * 8;             // 0..56

  for (int k0 = 0; k0 < K; k0 += 32) {
    // --- A tile 128x32 -> LDS via async DMA (2 x 16B per thread) ---
    {
      const u16* gp = A + (size_t)(bm + arow)*lda + k0 + 16*ahalf;
      async_copy_b128(aldsoff,      gp);
      async_copy_b128(aldsoff + 16, gp + 8);
    }
    // --- B tile 32x64 -> LDS transposed (1 x b128 load, 8 x b16 stores) ---
    {
      const uint4 q = *reinterpret_cast<const uint4*>(
          B + (size_t)(k0 + bk)*ldb + bn + bnn);
      const u16* e = reinterpret_cast<const u16*>(&q);
      #pragma unroll
      for (int j = 0; j < 8; ++j)
        Bs[(bnn + j)*40 + bk] = e[j];
    }
    if (k0 + 32 < K)  // prefetch next A tile -> global_prefetch_b8
      __builtin_prefetch(A + (size_t)(bm + arow)*lda + k0 + 32, 0, 1);

    wait_asynccnt0();          // this wave's async LDS writes landed
    __syncthreads();           // everyone's tile writes visible

    // --- fragments per documented 16-bit VGPR layouts ---
    FragAB af[2], bfr[2];
    #pragma unroll
    for (int i = 0; i < 2; ++i) {           // A 16x32: e0..7 <-> K=8h+0..7, e8..15 <-> K=16+8h+0..7
      int ml = wm*32 + i*16 + lane16;
      af[i].q[0] = *reinterpret_cast<const uint4*>(&As[ml*40 +      8*lh]);
      af[i].q[1] = *reinterpret_cast<const uint4*>(&As[ml*40 + 16 + 8*lh]);
    }
    #pragma unroll
    for (int j = 0; j < 2; ++j) {           // B 32x16: e0..7 <-> K=8h+0..7, e8..15 <-> K=16+8h+0..7
      int nl = wn*32 + j*16 + lane16;
      bfr[j].q[0] = *reinterpret_cast<const uint4*>(&Bs[nl*40 +      8*lh]);
      bfr[j].q[1] = *reinterpret_cast<const uint4*>(&Bs[nl*40 + 16 + 8*lh]);
    }
    #pragma unroll
    for (int i = 0; i < 2; ++i)
      #pragma unroll
      for (int j = 0; j < 2; ++j)
        acc[i][j] = __builtin_amdgcn_wmma_f32_16x16x32_bf16(
            false, af[i].v, false, bfr[j].v, (short)0, acc[i][j], false, false);
    __syncthreads();
  }

  // --- store: C 16x16 layout: lane<16 -> M=r, lane>=16 -> M=8+r; N = lane&15 ---
  #pragma unroll
  for (int i = 0; i < 2; ++i)
    #pragma unroll
    for (int j = 0; j < 2; ++j)
      #pragma unroll
      for (int r = 0; r < 8; ++r) {
        int row = bm + wm*32 + i*16 + lh*8 + r;
        int col = bn + wn*32 + j*16 + lane16;
        if constexpr (OUT_BF16)
          ((u16*)Cout)[(size_t)row*ldb + col] = f2bf(acc[i][j][r]);
        else
          ((float*)Cout)[(size_t)row*ldb + col] = acc[i][j][r];
      }
}

// ---------------- RMSNorm -> bf16, K padded to 1024 with zeros ----------------
__global__ __launch_bounds__(256)
void k_rmsnorm(const float* __restrict__ x, const float* __restrict__ w,
               u16* __restrict__ xn) {
  __shared__ float red[256];
  int row = blockIdx.x;
  const float* xr = x + (size_t)row * D_MODEL;
  float v[4]; float ss = 0.f;
  #pragma unroll
  for (int i = 0; i < 4; ++i) {
    int d = threadIdx.x + i*256;
    float t = (d < D_MODEL) ? xr[d] : 0.f;
    v[i] = t; ss += t*t;
  }
  red[threadIdx.x] = ss; __syncthreads();
  for (int s = 128; s > 0; s >>= 1) {
    if (threadIdx.x < s) red[threadIdx.x] += red[threadIdx.x + s];
    __syncthreads();
  }
  float rstd = rsqrtf(red[0] / (float)D_MODEL + 1e-5f);
  u16* out = xn + (size_t)row * D_MODEL_P;
  #pragma unroll
  for (int i = 0; i < 4; ++i) {
    int d = threadIdx.x + i*256;
    out[d] = (d < D_MODEL) ? f2bf(v[i]*rstd*w[d]) : (u16)0;
  }
}

// ------- fp32 weights [rows,cols] -> bf16 [rows_pad,cols_pad], zero padded -------
__global__ void k_cvt_pad(const float* __restrict__ W, u16* __restrict__ Wb,
                          int rows, int cols, int rows_pad, int cols_pad) {
  size_t idx = (size_t)blockIdx.x*blockDim.x + threadIdx.x;
  size_t total = (size_t)rows_pad * cols_pad;
  if (idx >= total) return;
  int r = (int)(idx / cols_pad); int c = (int)(idx % cols_pad);
  Wb[idx] = (r < rows && c < cols) ? f2bf(W[(size_t)r*cols + c]) : (u16)0;
}

// ---------------- depthwise causal conv (K=4) + SiLU ----------------
__global__ void k_conv_silu(const u16* __restrict__ xz, const float* __restrict__ cw,
                            const float* __restrict__ cb, float* __restrict__ xc) {
  size_t idx = (size_t)blockIdx.x*blockDim.x + threadIdx.x;
  if (idx >= (size_t)MROWS * D_INNER) return;
  int e = (int)(idx % D_INNER);
  int m = (int)(idx / D_INNER);
  int b = m / SEQ, l = m % SEQ;
  float acc = cb[e];
  #pragma unroll
  for (int j = 0; j < D_CONV; ++j) {
    int ll = l - (D_CONV-1) + j;
    if (ll >= 0)
      acc += bf2f(xz[(size_t)(b*SEQ + ll)*N_IN_P + e]) * cw[e*D_CONV + j];
  }
  xc[idx] = siluf_(acc);
}

// ---------------- xproj: dbc[m, 79] = xc[m, :] @ W_xproj ----------------
__global__ __launch_bounds__(128)
void k_xproj(const float* __restrict__ xc, const float* __restrict__ Wx,
             float* __restrict__ dbc) {
  __shared__ float sx[D_INNER];
  int m = blockIdx.x;
  const float* xr = xc + (size_t)m * D_INNER;
  for (int d = threadIdx.x; d < D_INNER; d += 128) sx[d] = xr[d];
  __syncthreads();
  int c = threadIdx.x;
  if (c < DBC_W) {
    float acc = 0.f;
    for (int k = 0; k < D_INNER; ++k) acc += sx[k] * Wx[(size_t)k*DBC_W + c];
    dbc[(size_t)m*DBC_W + c] = acc;
  }
}

// ---------------- delta = softplus(dt @ W_dt + b_dt) ----------------
__global__ __launch_bounds__(256)
void k_dt(const float* __restrict__ dbc, const float* __restrict__ Wdt,
          const float* __restrict__ bdt, float* __restrict__ delta) {
  __shared__ float sdt[DT_RANK];
  int m = blockIdx.x >> 3;
  int chunk = blockIdx.x & 7;
  if (threadIdx.x < DT_RANK) sdt[threadIdx.x] = dbc[(size_t)m*DBC_W + threadIdx.x];
  __syncthreads();
  int e = chunk*256 + threadIdx.x;
  if (e < D_INNER) {
    float acc = bdt[e];
    for (int k = 0; k < DT_RANK; ++k) acc += sdt[k]*Wdt[(size_t)k*D_INNER + e];
    delta[(size_t)m*D_INNER + e] = softplusf_(acc);
  }
}

// ---------------- selective scan, fused +Dp*x, *silu(z), bf16 out (K padded 2048) ----
__global__ __launch_bounds__(256)
void k_scan(const float* __restrict__ delta, const float* __restrict__ xc,
            const float* __restrict__ dbc, const u16* __restrict__ xz,
            const float* __restrict__ A_log, const float* __restrict__ Dp,
            u16* __restrict__ g) {
  int b = blockIdx.x >> 3;
  int e = (blockIdx.x & 7)*256 + threadIdx.x;     // 0..2047 (pad lanes write zeros)
  bool active = e < D_INNER;
  float a[D_STATE], h[D_STATE];
  float dp = 0.f;
  #pragma unroll
  for (int n = 0; n < D_STATE; ++n) { h[n] = 0.f; a[n] = 0.f; }
  if (active) {
    #pragma unroll
    for (int n = 0; n < D_STATE; ++n) a[n] = -__expf(A_log[e*D_STATE + n]);
    dp = Dp[e];
  }
  for (int l = 0; l < SEQ; ++l) {
    size_t row = (size_t)b*SEQ + l;
    u16 gv = 0;
    if (active) {
      float d  = delta[row*D_INNER + e];
      float xv = xc[row*D_INNER + e];
      const float* bc = dbc + row*DBC_W;
      float y = 0.f;
      #pragma unroll
      for (int n = 0; n < D_STATE; ++n) {
        float Bn = bc[DT_RANK + n];
        float Cn = bc[DT_RANK + D_STATE + n];
        h[n] = __expf(d*a[n])*h[n] + (d*Bn)*xv;
        y += h[n]*Cn;
      }
      y += dp*xv;
      float z = bf2f(xz[row*N_IN_P + D_INNER + e]);
      gv = f2bf(y * siluf_(z));
    }
    g[row*D_INNER_P + e] = gv;
  }
}

// ---------------- e[b,d] = mean_l (x + P) ----------------
__global__ __launch_bounds__(256)
void k_reduce_e(const float* __restrict__ x, const float* __restrict__ P,
                float* __restrict__ eb) {
  int b = blockIdx.x >> 2;
  int d = (blockIdx.x & 3)*256 + threadIdx.x;
  if (d >= D_MODEL) return;
  float acc = 0.f;
  for (int l = 0; l < SEQ; ++l) {
    size_t row = (size_t)b*SEQ + l;
    acc += x[row*D_MODEL + d] + P[row*N_OUT_P + d];
  }
  eb[b*D_MODEL + d] = acc * (1.f/(float)SEQ);
}

// ---------------- h = elu(tanh(e @ W_fc + b_fc)) ----------------
__global__ __launch_bounds__(256)
void k_fc(const float* __restrict__ eb, const float* __restrict__ Wfc,
          const float* __restrict__ bfc, float* __restrict__ outh) {
  __shared__ float se[D_MODEL];
  int b = blockIdx.x;
  for (int d = threadIdx.x; d < D_MODEL; d += 256) se[d] = eb[b*D_MODEL + d];
  __syncthreads();
  #pragma unroll
  for (int r = 0; r < 4; ++r) {
    int d = threadIdx.x + r*256;
    if (d < D_MODEL) {
      float acc = bfc[d];
      for (int k = 0; k < D_MODEL; ++k) acc += se[k]*Wfc[(size_t)k*D_MODEL + d];
      outh[b*D_MODEL + d] = eluf_(tanhf(acc));
    }
  }
}

// ---------------- mu / sigma heads ----------------
__global__ __launch_bounds__(128)
void k_head(const float* __restrict__ outh, const float* __restrict__ Wmu,
            const float* __restrict__ bmu, const float* __restrict__ Wsig,
            const float* __restrict__ bsig, float* __restrict__ outmu,
            float* __restrict__ outsig) {
  __shared__ float sh[D_MODEL];
  int b = blockIdx.x;
  for (int d = threadIdx.x; d < D_MODEL; d += 128) sh[d] = outh[b*D_MODEL + d];
  __syncthreads();
  int c = threadIdx.x & 63;
  const float* W = (threadIdx.x < 64) ? Wmu : Wsig;
  float acc = 0.f;
  for (int k = 0; k < D_MODEL; ++k) acc += sh[k]*W[(size_t)k*DIM_OUT + c];
  if (threadIdx.x < 64) outmu[b*DIM_OUT + c]  = acc + bmu[c];
  else                  outsig[b*DIM_OUT + c] = eluf_(acc + bsig[c]) + 1.f + 1e-14f;
}

// ---------------- launch ----------------
extern "C" void kernel_launch(void* const* d_in, const int* in_sizes, int n_in,
                              void* d_out, int out_size, void* d_ws, size_t ws_size,
                              hipStream_t stream) {
  (void)in_sizes; (void)n_in; (void)out_size; (void)ws_size;
  const float* x      = (const float*)d_in[0];
  const float* w_norm = (const float*)d_in[1];
  const float* W_in   = (const float*)d_in[2];
  const float* conv_w = (const float*)d_in[3];
  const float* conv_b = (const float*)d_in[4];
  const float* W_xprj = (const float*)d_in[5];
  const float* W_dt   = (const float*)d_in[6];
  const float* b_dt   = (const float*)d_in[7];
  const float* A_log  = (const float*)d_in[8];
  const float* Dp     = (const float*)d_in[9];
  const float* W_outp = (const float*)d_in[10];
  const float* W_fc   = (const float*)d_in[11];
  const float* b_fc   = (const float*)d_in[12];
  const float* W_mu   = (const float*)d_in[13];
  const float* b_mu   = (const float*)d_in[14];
  const float* W_sig  = (const float*)d_in[15];
  const float* b_sig  = (const float*)d_in[16];
  float* out = (float*)d_out;

  char* ws = (char*)d_ws;
  size_t off = 0;
  auto alloc = [&](size_t bytes) {
    char* p = ws + off; off += (bytes + 255) & ~(size_t)255; return p;
  };
  u16*   xn    = (u16*)  alloc((size_t)MROWS*D_MODEL_P*2);   // 32 MB
  u16*   Winb  = (u16*)  alloc((size_t)D_MODEL_P*N_IN_P*2);  // 8.3 MB
  u16*   Woutb = (u16*)  alloc((size_t)D_INNER_P*N_OUT_P*2); // 4.2 MB
  u16*   xz    = (u16*)  alloc((size_t)MROWS*N_IN_P*2);      // 132 MB
  float* xc    = (float*)alloc((size_t)MROWS*D_INNER*4);     // 125 MB
  float* dbc   = (float*)alloc((size_t)MROWS*DBC_W*4);       // 5 MB
  float* delta = (float*)alloc((size_t)MROWS*D_INNER*4);     // 125 MB
  u16*   g     = (u16*)  alloc((size_t)MROWS*D_INNER_P*2);   // 64 MB
  float* eb    = (float*)alloc((size_t)BATCH*D_MODEL*4);
  float* P     = delta;  // alias: delta dead after k_scan; P needs 64 MB <= 125 MB

  k_rmsnorm<<<MROWS, 256, 0, stream>>>(x, w_norm, xn);
  {
    size_t tot = (size_t)D_MODEL_P*N_IN_P;
    k_cvt_pad<<<(unsigned)((tot+255)/256), 256, 0, stream>>>(W_in, Winb, D_MODEL, N_IN, D_MODEL_P, N_IN_P);
  }
  {
    size_t tot = (size_t)D_INNER_P*N_OUT_P;
    k_cvt_pad<<<(unsigned)((tot+255)/256), 256, 0, stream>>>(W_outp, Woutb, D_INNER, D_MODEL, D_INNER_P, N_OUT_P);
  }
  // GEMM1: xz = xn @ W_in   (M=16384, Np=4032, K=1024)
  k_gemm_bf16<true ><<<dim3(N_IN_P/64,  MROWS/128), 256, 0, stream>>>(xn, Winb, xz, D_MODEL_P, D_MODEL_P, N_IN_P);
  {
    size_t tot = (size_t)MROWS*D_INNER;
    k_conv_silu<<<(unsigned)((tot+255)/256), 256, 0, stream>>>(xz, conv_w, conv_b, xc);
  }
  k_xproj<<<MROWS, 128, 0, stream>>>(xc, W_xprj, dbc);
  k_dt<<<MROWS*8, 256, 0, stream>>>(dbc, W_dt, b_dt, delta);
  k_scan<<<BATCH*8, 256, 0, stream>>>(delta, xc, dbc, xz, A_log, Dp, g);
  // GEMM4: P = g @ W_outp   (M=16384, Np=1024, K=2048)
  k_gemm_bf16<false><<<dim3(N_OUT_P/64, MROWS/128), 256, 0, stream>>>(g, Woutb, P, D_INNER_P, D_INNER_P, N_OUT_P);
  k_reduce_e<<<BATCH*4, 256, 0, stream>>>(x, P, eb);
  k_fc<<<BATCH, 256, 0, stream>>>(eb, W_fc, b_fc, out);
  k_head<<<BATCH, 128, 0, stream>>>(out, W_mu, b_mu, W_sig, b_sig,
                                    out + BATCH*D_MODEL,
                                    out + BATCH*D_MODEL + BATCH*DIM_OUT);
}
